// QD3DTrackGraph_39651138076960
// MI455X (gfx1250) — matrix operand order repaired and checked by reading, not compiled, not run on hardware
//
#include <hip/hip_runtime.h>
#include <hip/hip_bf16.h>
#include <math.h>

// ---------------------------------------------------------------- types
typedef _Float16 v16h __attribute__((ext_vector_type(16)));
typedef _Float16 v8h  __attribute__((ext_vector_type(8)));
typedef float    v8f  __attribute__((ext_vector_type(8)));

#define OBJ_SCORE_THR        0.5f
#define NMS_BACKDROP_IOU_THR 0.3f
#define NMS_CLASS_IOU_THR    0.7f
#define EPS_PD  1e-6f
#define EPS_COS 1e-8f

// ---------------------------------------------------------------- 1) rank -> perm
// Stable descending rank: rank[i] = #{j : s[j]>s[i]} + #{j<i : s[j]==s[i]}
__global__ void rank_kernel(const float* __restrict__ scores, int* __restrict__ perm, int n) {
    __shared__ float sc[2048];
    for (int t = threadIdx.x; t < n; t += blockDim.x) sc[t] = scores[t];
    __syncthreads();
    int i = blockIdx.x * blockDim.x + threadIdx.x;
    if (i < n) {
        float si = sc[i];
        int r = 0;
        for (int j = 0; j < n; ++j) {
            float sj = sc[j];
            r += (sj > si) || (sj == si && j < i);
        }
        perm[r] = i;
    }
}

// ---------------------------------------------------------------- 2) f32 -> f16 converts
__global__ void cvt_emb_kernel(const float* __restrict__ emb, const int* __restrict__ perm,
                               _Float16* __restrict__ out, int n, int k) {
    int idx = blockIdx.x * blockDim.x + threadIdx.x;
    if (idx < n * k) {
        int r = idx / k, c = idx - r * k;
        out[idx] = (_Float16)emb[(size_t)perm[r] * k + c];
    }
}

__global__ void cvt_memo_kernel(const float* __restrict__ src, _Float16* __restrict__ out, int total) {
    int idx = blockIdx.x * blockDim.x + threadIdx.x;
    if (idx < total) out[idx] = (_Float16)src[idx];
}

// ---------------------------------------------------------------- 3) NMS validity
__global__ void nms_kernel(const float* __restrict__ boxes, const float* __restrict__ scores,
                           const int* __restrict__ perm, int* __restrict__ valids, int n) {
    __shared__ int flag;
    int i = blockIdx.x;
    if (threadIdx.x == 0) flag = 0;
    __syncthreads();
    int pi = perm[i];
    float x1 = boxes[pi * 4 + 0], y1 = boxes[pi * 4 + 1];
    float x2 = boxes[pi * 4 + 2], y2 = boxes[pi * 4 + 3];
    float ai = (x2 - x1) * (y2 - y1);
    float thr = (scores[pi] < OBJ_SCORE_THR) ? NMS_BACKDROP_IOU_THR : NMS_CLASS_IOU_THR;
    int sup = 0;
    for (int j = threadIdx.x; j < i; j += blockDim.x) {
        int pj = perm[j];
        float bx1 = boxes[pj * 4 + 0], by1 = boxes[pj * 4 + 1];
        float bx2 = boxes[pj * 4 + 2], by2 = boxes[pj * 4 + 3];
        float aj = (bx2 - bx1) * (by2 - by1);
        float lx = fmaxf(x1, bx1), ly = fmaxf(y1, by1);
        float rx = fminf(x2, bx2), ry = fminf(y2, by2);
        float w = fmaxf(rx - lx, 0.f), h = fmaxf(ry - ly, 0.f);
        float inter = w * h;
        float uni = fmaxf(ai + aj - inter, 1e-7f);
        sup |= (inter / uni) > thr;
    }
    if (sup) atomicOr(&flag, 1);
    __syncthreads();
    if (threadIdx.x == 0) valids[i] = flag ? 0 : 1;
}

// ---------------------------------------------------------------- 4) WMMA GEMM: C[N,M] = A[N,K] * B[M,K]^T
// 4 waves/block; each wave computes a 32x32 tile as 2x2 WMMA 16x16 tiles.
__global__ __launch_bounds__(128)
void gemm_kernel(const _Float16* __restrict__ A, const _Float16* __restrict__ B,
                 float* __restrict__ C, int M, int K) {
    const int lane = threadIdx.x & 31;
    const int wave = threadIdx.x >> 5;
    const int row0 = blockIdx.y * 64 + (wave >> 1) * 32;
    const int col0 = blockIdx.x * 64 + (wave & 1) * 32;
    const int half = lane >> 4;   // lane group 0 / 1
    const int l16  = lane & 15;

    // A fragment k order used here (contiguous; ISA table order differs): lanes 0-15 hold K 0..15, lanes 16-31 hold K 16..31
    const int akoff_lo = half ? 16 : 0;
    const int akoff_hi = half ? 24 : 8;
    // B fragment k order used here (contiguous, matches A): lanes 0-15 hold K 0..15, lanes 16-31 hold K 16..31
    const int bkoff = half ? 16 : 0;

    const _Float16* arow = A + (size_t)(row0 + l16) * K;
    const _Float16* brow = B + (size_t)(col0 + l16) * K;

    v8f c00 = {}, c01 = {}, c10 = {}, c11 = {};
    for (int kk = 0; kk < K; kk += 32) {
        v8h a0lo = *(const v8h*)(arow + kk + akoff_lo);
        v8h a0hi = *(const v8h*)(arow + kk + akoff_hi);
        v8h a1lo = *(const v8h*)(arow + (size_t)16 * K + kk + akoff_lo);
        v8h a1hi = *(const v8h*)(arow + (size_t)16 * K + kk + akoff_hi);
        v16h a0 = __builtin_shufflevector(a0lo, a0hi, 0,1,2,3,4,5,6,7,8,9,10,11,12,13,14,15);
        v16h a1 = __builtin_shufflevector(a1lo, a1hi, 0,1,2,3,4,5,6,7,8,9,10,11,12,13,14,15);
        v16h b0 = *(const v16h*)(brow + kk + bkoff);
        v16h b1 = *(const v16h*)(brow + (size_t)16 * K + kk + bkoff);

        c00 = __builtin_amdgcn_wmma_f32_16x16x32_f16(false, a0, false, b0, (short)0, c00, false, false);
        c01 = __builtin_amdgcn_wmma_f32_16x16x32_f16(false, a0, false, b1, (short)0, c01, false, false);
        c10 = __builtin_amdgcn_wmma_f32_16x16x32_f16(false, a1, false, b0, (short)0, c10, false, false);
        c11 = __builtin_amdgcn_wmma_f32_16x16x32_f16(false, a1, false, b1, (short)0, c11, false, false);
    }

    // C/D layout: lanes 0-15 -> N=lane, M=v; lanes 16-31 -> N=lane-16, M=v+8
    const int mrow = half ? 8 : 0;
#pragma unroll
    for (int v = 0; v < 8; ++v) {
        C[(size_t)(row0 + mrow + v)      * M + (col0 + l16)]      = c00[v];
        C[(size_t)(row0 + mrow + v)      * M + (col0 + 16 + l16)] = c01[v];
        C[(size_t)(row0 + 16 + mrow + v) * M + (col0 + l16)]      = c10[v];
        C[(size_t)(row0 + 16 + mrow + v) * M + (col0 + 16 + l16)] = c11[v];
    }
}

// ---------------------------------------------------------------- 5) softmax stats
__device__ __forceinline__ void sm_combine(float& m, float& s, float m2, float s2) {
    float nm = fmaxf(m, m2);
    s = s * expf(m - nm) + s2 * expf(m2 - nm);
    m = nm;
}

__global__ void row_stats_kernel(const float* __restrict__ feats, float* __restrict__ rmax,
                                 float* __restrict__ rsum, int M) {
    __shared__ float sm[256], ss[256];
    int i = blockIdx.x;
    const float* rp = feats + (size_t)i * M;
    float m = -__builtin_inff(), s = 0.f;
    for (int j = threadIdx.x; j < M; j += blockDim.x) {
        float x = rp[j];
        float nm = fmaxf(m, x);
        s = s * expf(m - nm) + expf(x - nm);
        m = nm;
    }
    sm[threadIdx.x] = m; ss[threadIdx.x] = s;
    __syncthreads();
    for (int off = 128; off > 0; off >>= 1) {
        if (threadIdx.x < off) {
            float m2 = sm[threadIdx.x + off], s2 = ss[threadIdx.x + off];
            float mm = sm[threadIdx.x], sss = ss[threadIdx.x];
            sm_combine(mm, sss, m2, s2);
            sm[threadIdx.x] = mm; ss[threadIdx.x] = sss;
        }
        __syncthreads();
    }
    if (threadIdx.x == 0) { rmax[i] = sm[0]; rsum[i] = ss[0]; }
}

__global__ void col_stats_partial_kernel(const float* __restrict__ feats, float* __restrict__ pmax,
                                         float* __restrict__ psum, int rowsPerBlk, int M) {
    int col = blockIdx.x * blockDim.x + threadIdx.x;
    int r0 = blockIdx.y * rowsPerBlk;
    float m = -__builtin_inff(), s = 0.f;
    for (int i = r0; i < r0 + rowsPerBlk; ++i) {
        float x = feats[(size_t)i * M + col];
        float nm = fmaxf(m, x);
        s = s * expf(m - nm) + expf(x - nm);
        m = nm;
    }
    pmax[(size_t)blockIdx.y * M + col] = m;
    psum[(size_t)blockIdx.y * M + col] = s;
}

__global__ void col_stats_combine_kernel(const float* __restrict__ pmax, const float* __restrict__ psum,
                                         float* __restrict__ cmax, float* __restrict__ csum, int M, int P) {
    int col = blockIdx.x * blockDim.x + threadIdx.x;
    float m = -__builtin_inff(), s = 0.f;
    for (int p = 0; p < P; ++p)
        sm_combine(m, s, pmax[(size_t)p * M + col], psum[(size_t)p * M + col]);
    cmax[col] = m; csum[col] = s;
}

// ---------------------------------------------------------------- 6) final fused affinity (in-place on feats)
__global__ void final_kernel(float* __restrict__ out,
                             const float* __restrict__ obs_3d, const int* __restrict__ perm,
                             const int* __restrict__ valids,
                             const float* __restrict__ memo_pred, const float* __restrict__ memo_obs,
                             const float* __restrict__ memo_vs,
                             const int* __restrict__ det_cls, const int* __restrict__ memo_cls,
                             const float* __restrict__ rmax, const float* __restrict__ rsum,
                             const float* __restrict__ cmax, const float* __restrict__ csum,
                             int M) {
    int i = blockIdx.y;
    int j = blockIdx.x * blockDim.x + threadIdx.x;
    int pi = perm[i];
    const float* op = obs_3d + (size_t)pi * 7;
    float o[7];
#pragma unroll
    for (int d = 0; d < 7; ++d) o[d] = op[d];
    float ro = rmax[i], rs = rsum[i];
    int   ci = det_cls[pi];
    int   vi = valids[i];

    const float* pp = memo_pred + (size_t)j * 7;
    const float* mo = memo_obs  + (size_t)j * 7;
    const float* mv = memo_vs   + (size_t)j * 7;

    // 7-dim affinity
    float d7 = 0.f;
#pragma unroll
    for (int d = 0; d < 7; ++d) { float df = o[d] - pp[d] + EPS_PD; d7 += df * df; }
    float s_iou = expf(-sqrtf(d7) / 10.0f);

    // centroid weight (3-dim)
    float d3 = 0.f;
#pragma unroll
    for (int d = 0; d < 3; ++d) { float df = o[d] - pp[d] + EPS_PD; d3 += df * df; }
    float cw = expf(-sqrtf(d3) / 10.0f);

    // motion weight
    float m2 = 0.f;
#pragma unroll
    for (int d = 0; d < 3; ++d) { float md = (o[d] - mo[d]) - mv[d] + EPS_PD; m2 += md * md; }
    float mw = expf(-sqrtf(m2) / 5.0f);

    // cosine between observation direction and memo velocity (2D)
    float dx = o[0] - mo[0], dy = o[1] - mo[1];
    float dot = dx * mv[0] + dy * mv[1];
    float na = fmaxf(sqrtf(dx * dx + dy * dy), EPS_COS);
    float nb = fmaxf(sqrtf(mv[0] * mv[0] + mv[1] * mv[1]), EPS_COS);
    float cosv = (dot / (na * nb) + 1.0f) * 0.5f;
    float s_depth = cosv * cw + (1.0f - cosv) * mw;

    // bi-softmax embedding similarity
    size_t idx = (size_t)i * M + j;
    float f = out[idx];
    float sim = 0.5f * (expf(f - ro) / rs + expf(f - cmax[j]) / csum[j]);

    float s = 0.5f * s_iou * s_depth + 0.5f * sim;  // / (0.5 + 0.5)
    s *= (s_iou   > 0.f) ? 1.f : 0.f;
    s *= (s_depth > 0.f) ? 1.f : 0.f;
    s *= (ci == memo_cls[j]) ? 1.f : 0.f;
    s *= vi ? 1.f : 0.f;
    out[idx] = s;
}

// ---------------------------------------------------------------- launch
extern "C" void kernel_launch(void* const* d_in, const int* in_sizes, int n_in,
                              void* d_out, int out_size, void* d_ws, size_t ws_size,
                              hipStream_t stream) {
    const float* det_boxes   = (const float*)d_in[0];
    const float* det_scores  = (const float*)d_in[1];
    const float* obs_3d      = (const float*)d_in[2];
    const float* embeddings  = (const float*)d_in[3];
    const float* memo_obs    = (const float*)d_in[4];
    const float* memo_pred   = (const float*)d_in[5];
    const float* memo_vs     = (const float*)d_in[6];
    const float* memo_embeds = (const float*)d_in[7];
    const int*   det_cls     = (const int*)d_in[8];
    const int*   memo_cls    = (const int*)d_in[9];

    const int N = in_sizes[1];         // 2048 detections
    const int M = in_sizes[9];         // 4096 memo tracks
    const int E = in_sizes[3] / N;     // 512 embedding dims
    float* out = (float*)d_out;        // feats staged here, then rewritten in place

    // ---- workspace carve-up (all chunks 16B aligned) ----
    char* w = (char*)d_ws;
    int*   perm   = (int*)w;    w += (size_t)N * 4;
    int*   valids = (int*)w;    w += (size_t)N * 4;
    float* rmax   = (float*)w;  w += (size_t)N * 4;
    float* rsum   = (float*)w;  w += (size_t)N * 4;
    float* cmax   = (float*)w;  w += (size_t)M * 4;
    float* csum   = (float*)w;  w += (size_t)M * 4;
    const int P = 8;
    float* pmax   = (float*)w;  w += (size_t)P * M * 4;
    float* psum   = (float*)w;  w += (size_t)P * M * 4;
    _Float16* emb16  = (_Float16*)w;  w += (size_t)N * E * 2;
    _Float16* memo16 = (_Float16*)w;

    // 1) stable descending argsort via counting rank
    rank_kernel<<<(N + 255) / 256, 256, 0, stream>>>(det_scores, perm, N);

    // 2) f16 copies (embeddings gathered through perm)
    cvt_emb_kernel<<<((size_t)N * E + 255) / 256, 256, 0, stream>>>(embeddings, perm, emb16, N, E);
    cvt_memo_kernel<<<((size_t)M * E + 255) / 256, 256, 0, stream>>>(memo_embeds, memo16, M * E);

    // 3) cross-class NMS validity
    nms_kernel<<<N, 256, 0, stream>>>(det_boxes, det_scores, perm, valids, N);

    // 4) WMMA GEMM: feats = emb_sorted(f16) @ memo(f16)^T  -> d_out (f32)
    dim3 gg(M / 64, N / 64);
    gemm_kernel<<<gg, 128, 0, stream>>>(emb16, memo16, out, M, E);

    // 5) softmax statistics over rows (axis=1) and columns (axis=0)
    row_stats_kernel<<<N, 256, 0, stream>>>(out, rmax, rsum, M);
    dim3 gc(M / 256, P);
    col_stats_partial_kernel<<<gc, 256, 0, stream>>>(out, pmax, psum, N / P, M);
    col_stats_combine_kernel<<<M / 256, 256, 0, stream>>>(pmax, psum, cmax, csum, M, P);

    // 6) fused affinity, in place over d_out
    dim3 gf(M / 256, N);
    final_kernel<<<gf, 256, 0, stream>>>(out, obs_3d, perm, valids, memo_pred, memo_obs,
                                         memo_vs, det_cls, memo_cls, rmax, rsum, cmax, csum, M);
}
